// MultiHeadSelfAttention_45466523795534
// MI455X (gfx1250) — matrix operand order repaired and checked by reading, not compiled, not run on hardware
//
#include <hip/hip_runtime.h>
#include <stdint.h>

#define BB 16
#define NN 577
#define DD 768
#define HH 12
#define HDIM 64
#define MM (BB * NN)   // 9232 total tokens
#define NPAD 608       // 19*32, padded token count for V^T / P*V K-dim
#define SSTRIDE 609    // fp32 score row stride (odd dwords -> low bank conflict)
#define PSTRIDE 616    // bf16 prob row stride (1232B = multiple of 16B for b128 ds loads)

typedef __attribute__((ext_vector_type(16))) __bf16 v16bf;
typedef __attribute__((ext_vector_type(8))) float v8f;
typedef __attribute__((ext_vector_type(8))) unsigned short u16x8;

union Frag16 { v16bf v; unsigned short u[16]; };

__device__ __forceinline__ unsigned short f2bf(float f) {
  union { float f; unsigned int u; } a; a.f = f;
  unsigned int r = a.u + 0x7FFFu + ((a.u >> 16) & 1u);  // RNE
  return (unsigned short)(r >> 16);
}

// Build a 16-element bf16 WMMA fragment from two contiguous 8-element (16B) chunks.
__device__ __forceinline__ v16bf ldfrag(const unsigned short* p0, const unsigned short* p1) {
  u16x8 lo = *(const u16x8*)p0;
  u16x8 hi = *(const u16x8*)p1;
  Frag16 f;
#pragma unroll
  for (int i = 0; i < 8; ++i) { f.u[i] = lo[i]; f.u[i + 8] = hi[i]; }
  return f.v;
}

__global__ void conv_x_kernel(const float* __restrict__ x, unsigned short* __restrict__ xb, int n) {
  int i = blockIdx.x * 256 + threadIdx.x;
  if (i < n) xb[i] = f2bf(x[i]);
}

__global__ void conv_w_kernel(const float* __restrict__ wq, const float* __restrict__ wk,
                              const float* __restrict__ wv, unsigned short* __restrict__ tq,
                              unsigned short* __restrict__ tk, unsigned short* __restrict__ tv) {
  int i = blockIdx.x * 256 + threadIdx.x;
  if (i < DD * DD) {
    int k = i / DD, n = i - k * DD;
    size_t o = (size_t)n * DD + k;  // store transposed: Wt[n][k]
    tq[o] = f2bf(wq[i]); tk[o] = f2bf(wk[i]); tv[o] = f2bf(wv[i]);
  }
}

__global__ void zero_kernel(unsigned int* __restrict__ p, int n) {
  int i = blockIdx.x * 256 + threadIdx.x;
  if (i < n) p[i] = 0u;
}

// C = x @ W + bias ; 8 waves per block, each wave does a 16x64 tile via 4 accumulators.
// mode 0/1: store bf16 row-major [MM][DD] (Q,K). mode 2: store V transposed [BB*DD][NPAD].
__global__ __launch_bounds__(256) void qkv_gemm_kernel(
    const unsigned short* __restrict__ xb, const unsigned short* __restrict__ wt,
    const float* __restrict__ bias, unsigned short* __restrict__ outRM,
    unsigned short* __restrict__ outVt, int mode) {
  const int lane = threadIdx.x & 31, wave = threadIdx.x >> 5;
  const int lh = lane >> 4, ll = lane & 15;
  const int rowBase = blockIdx.x * 128 + wave * 16;
  const int colBase = blockIdx.y * 64;

  v8f acc[4];
#pragma unroll
  for (int i = 0; i < 4; ++i) acc[i] = {};

  const int tokA = (rowBase + ll < MM) ? (rowBase + ll) : (MM - 1);
  const unsigned short* arow = xb + (size_t)tokA * DD;

  for (int ks = 0; ks < DD; ks += 32) {
    v16bf a = ldfrag(arow + ks + 8 * lh, arow + ks + 8 * lh + 16);
#pragma unroll
    for (int nt = 0; nt < 4; ++nt) {
      const int n = colBase + nt * 16 + ll;
      const unsigned short* brow = wt + (size_t)n * DD + ks + 8 * lh;
      v16bf bm = ldfrag(brow, brow + 16);
      acc[nt] = __builtin_amdgcn_wmma_f32_16x16x32_bf16(false, a, false, bm, (short)0,
                                                        acc[nt], false, false);
    }
  }

#pragma unroll
  for (int nt = 0; nt < 4; ++nt) {
    const int n = colBase + nt * 16 + ll;
    const float bvs = bias[n];
#pragma unroll
    for (int r = 0; r < 8; ++r) {
      const int m = rowBase + r + 8 * lh;
      if (m < MM) {
        const float v = acc[nt][r] + bvs;
        if (mode < 2) {
          outRM[(size_t)m * DD + n] = f2bf(v);
        } else {
          const int bidx = m / NN;
          const int tok = m - bidx * NN;
          outVt[((size_t)bidx * DD + n) * NPAD + tok] = f2bf(v);
        }
      }
    }
  }
}

// One block (4 waves) per (b, h, 16-query strip): scores -> softmax -> attn write -> P@V.
__global__ __launch_bounds__(128) void attention_kernel(
    const unsigned short* __restrict__ qb, const unsigned short* __restrict__ kb,
    const unsigned short* __restrict__ vt, float* __restrict__ outO,
    float* __restrict__ outA) {
  __shared__ float sS[16 * SSTRIDE];          // fp32 scores -> exp values
  __shared__ unsigned short sP[16 * PSTRIDE]; // bf16 exp values (WMMA A operand)
  __shared__ float sRed[16 * 8];
  __shared__ float sMax[16];
  __shared__ float sRinv[16];

  const int qt = blockIdx.x, h = blockIdx.y, b = blockIdx.z;
  const int tid = threadIdx.x;
  const int lane = tid & 31, wave = tid >> 5;
  const int lh = lane >> 4, ll = lane & 15;
  const float scale = 0.03608439182435161f;  // 1/sqrt(768): module scales by embed_dim

  // Q fragments for this 16-row strip (two 32-wide K chunks of head_dim=64).
  const int qtok = (qt * 16 + ll < NN) ? (qt * 16 + ll) : (NN - 1);
  const unsigned short* qrow = qb + (size_t)(b * NN + qtok) * DD + h * HDIM;
  v16bf aQ0 = ldfrag(qrow + 8 * lh, qrow + 8 * lh + 16);
  v16bf aQ1 = ldfrag(qrow + 32 + 8 * lh, qrow + 32 + 8 * lh + 16);

  // Pass 1: scaled scores into LDS. 37 key tiles round-robined over 4 waves.
  for (int ct = wave; ct < 37; ct += 4) {
    const int ktok = (ct * 16 + ll < NN) ? (ct * 16 + ll) : (NN - 1);
    const unsigned short* krow = kb + (size_t)(b * NN + ktok) * DD + h * HDIM;
    v8f acc = {};
    v16bf b0 = ldfrag(krow + 8 * lh, krow + 8 * lh + 16);
    acc = __builtin_amdgcn_wmma_f32_16x16x32_bf16(false, aQ0, false, b0, (short)0, acc, false, false);
    v16bf b1 = ldfrag(krow + 32 + 8 * lh, krow + 32 + 8 * lh + 16);
    acc = __builtin_amdgcn_wmma_f32_16x16x32_bf16(false, aQ1, false, b1, (short)0, acc, false, false);
#pragma unroll
    for (int r = 0; r < 8; ++r)
      sS[(r + 8 * lh) * SSTRIDE + ct * 16 + ll] = acc[r] * scale;
  }
  __syncthreads();

  // Softmax statistics: 8 threads per row.
  const int row = tid >> 3, sub = tid & 7;
  float pmax = -3.0e38f;
  for (int c = sub; c < NN; c += 8) pmax = fmaxf(pmax, sS[row * SSTRIDE + c]);
  sRed[row * 8 + sub] = pmax;
  __syncthreads();
  if (tid < 16) {
    float m = sRed[tid * 8];
#pragma unroll
    for (int i = 1; i < 8; ++i) m = fmaxf(m, sRed[tid * 8 + i]);
    sMax[tid] = m;
  }
  __syncthreads();
  const float rmax = sMax[row];
  float psum = 0.f;
  for (int c = sub; c < NPAD; c += 8) {
    float e = 0.f;
    if (c < NN) {
      e = __expf(sS[row * SSTRIDE + c] - rmax);
      sS[row * SSTRIDE + c] = e;
      psum += e;
    }
    sP[row * PSTRIDE + c] = f2bf(e);  // zero-pads columns 577..607
  }
  sRed[row * 8 + sub] = psum;
  __syncthreads();
  if (tid < 16) {
    float s = 0.f;
#pragma unroll
    for (int i = 0; i < 8; ++i) s += sRed[tid * 8 + i];
    sRinv[tid] = 1.0f / s;
  }
  __syncthreads();

  // Write attention probabilities (normalized, fp32, fully coalesced).
  for (int r = 0; r < 16; ++r) {
    const int t = qt * 16 + r;
    if (t >= NN) break;
    const float rinv = sRinv[r];
    float* dst = outA + ((size_t)(b * HH + h) * NN + t) * NN;
    for (int c = tid; c < NN; c += 128) dst[c] = sS[r * SSTRIDE + c] * rinv;
  }

  // Pass 2: out_tile = P @ V. Each wave owns one 16-dim slice; 1/sum folded into epilogue.
  {
    const int dim = wave * 16 + ll;
    v8f acc = {};
    const unsigned short* vbase = vt + ((size_t)b * DD + h * HDIM + dim) * NPAD;
    for (int kt = 0; kt < 19; ++kt) {
      const unsigned short* prow = &sP[ll * PSTRIDE + kt * 32 + 8 * lh];
      v16bf a = ldfrag(prow, prow + 16);
      const unsigned short* vrow = vbase + kt * 32 + 8 * lh;
      v16bf bv = ldfrag(vrow, vrow + 16);
      acc = __builtin_amdgcn_wmma_f32_16x16x32_bf16(false, a, false, bv, (short)0, acc, false, false);
    }
#pragma unroll
    for (int r = 0; r < 8; ++r) {
      const int rr = r + 8 * lh;
      const int t = qt * 16 + rr;
      if (t < NN)
        outO[((size_t)(b * NN + t)) * DD + h * HDIM + dim] = acc[r] * sRinv[rr];
    }
  }
}

extern "C" void kernel_launch(void* const* d_in, const int* in_sizes, int n_in,
                              void* d_out, int out_size, void* d_ws, size_t ws_size,
                              hipStream_t stream) {
  (void)in_sizes; (void)n_in; (void)out_size; (void)ws_size;
  const float* x  = (const float*)d_in[0];
  const float* Wq = (const float*)d_in[1];
  const float* bq = (const float*)d_in[2];
  const float* Wk = (const float*)d_in[3];
  const float* bk = (const float*)d_in[4];
  const float* Wv = (const float*)d_in[5];
  const float* bv = (const float*)d_in[6];
  float* out  = (float*)d_out;
  float* attn = out + (size_t)BB * NN * DD;

  // Workspace carve (bf16 elements): ~58 MB total.
  unsigned short* ws  = (unsigned short*)d_ws;
  unsigned short* xb  = ws;                       // MM*DD
  unsigned short* wtq = xb + (size_t)MM * DD;     // DD*DD each, transposed
  unsigned short* wtk = wtq + (size_t)DD * DD;
  unsigned short* wtv = wtk + (size_t)DD * DD;
  unsigned short* qb  = wtv + (size_t)DD * DD;    // MM*DD
  unsigned short* kb  = qb + (size_t)MM * DD;     // MM*DD
  unsigned short* vt  = kb + (size_t)MM * DD;     // BB*DD*NPAD

  { int n = MM * DD; conv_x_kernel<<<(n + 255) / 256, 256, 0, stream>>>(x, xb, n); }
  { int n = DD * DD; conv_w_kernel<<<(n + 255) / 256, 256, 0, stream>>>(Wq, Wk, Wv, wtq, wtk, wtv); }
  { int n = (BB * DD * NPAD) / 2; zero_kernel<<<(n + 255) / 256, 256, 0, stream>>>((unsigned int*)vt, n); }

  dim3 gg((MM + 127) / 128, DD / 64);
  qkv_gemm_kernel<<<gg, 256, 0, stream>>>(xb, wtq, bq, qb, nullptr, 0);
  qkv_gemm_kernel<<<gg, 256, 0, stream>>>(xb, wtk, bk, kb, nullptr, 1);
  qkv_gemm_kernel<<<gg, 256, 0, stream>>>(xb, wtv, bv, nullptr, vt, 2);

  dim3 ga(37, HH, BB);
  attention_kernel<<<ga, 128, 0, stream>>>(qb, kb, vt, out, attn);
}
